// CrossAttentionBlock_83133386981931
// MI455X (gfx1250) — compile-verified, matrix operand order checked by reading, not run on hardware
//
#include <hip/hip_runtime.h>
#include <hip/hip_bf16.h>
#include <math.h>

#define C_  256
#define H_  8
#define D_  32
#define T1_ 2048
#define T2_ 2048
#define B_  2
#define R_  (B_ * T1_)   // 4096 rows (B*T)

typedef __bf16 bf16_t;
typedef bf16_t v16bf __attribute__((ext_vector_type(16)));
typedef bf16_t v8bf  __attribute__((ext_vector_type(8)));
typedef float  v8f   __attribute__((ext_vector_type(8)));

// WMMA 16x16x32 bf16 fragment: lane's 16 elements are two contiguous 8-element
// spans of K:  [kk + half*8 .. +7]  and  [kk + 16 + half*8 .. +7]
__device__ __forceinline__ v16bf frag_ld(const bf16_t* p, int kk, int half) {
  union { v16bf v; v8bf h[2]; } u;
  u.h[0] = *(const v8bf*)(p + kk + half * 8);
  u.h[1] = *(const v8bf*)(p + kk + 16 + half * 8);
  return u.v;
}
__device__ __forceinline__ v8f wmma_bf16(v16bf a, v16bf b, v8f c) {
  return __builtin_amdgcn_wmma_f32_16x16x32_bf16(false, a, false, b, (short)0, c,
                                                 false, false);
}

// =====================================================================
// Weight prep: Wt[n*K + k] = bf16(W[k*N + n])  (transpose + convert once)
// =====================================================================
__global__ void cab_w2bf_t(const float* __restrict__ W, bf16_t* __restrict__ Wt,
                           int K, int N) {
  int idx = blockIdx.x * blockDim.x + threadIdx.x;
  if (idx >= K * N) return;
  int k = idx / N, n = idx - k * N;          // coalesced read over n
  Wt[(size_t)n * K + k] = (bf16_t)W[idx];
}

// =====================================================================
// Kernel 1: row LayerNorm then GEMM  out[r,n] = act((LN(X)@W + bias)*scale)
// X: [R,256] f32, Wt: [N,256] bf16 (transposed), out: [R,N] bf16.
// block=128 (4 waves); each wave computes a 16x64 strip (4 WMMA tiles,
// A fragment loaded once per k-step and reused 4x).
// grid = (R/16, N/256)
// =====================================================================
__global__ void cab_ln_gemm(const float* __restrict__ X,
                            const float* __restrict__ gamma,
                            const float* __restrict__ beta,
                            const bf16_t* __restrict__ Wt,
                            const float* __restrict__ bias,
                            bf16_t* __restrict__ out,
                            int N, float scale, int gelu) {
  const int tid  = threadIdx.x;
  const int lane = tid & 31;
  const int wave = tid >> 5;
  const int half = lane >> 4;
  const int row0 = blockIdx.x * 16;
  const int col0 = blockIdx.y * 256 + wave * 64;

  __shared__ float  xr[16][C_];
  __shared__ float  red[16][8][2];
  __shared__ float  mu[16], rs[16];
  __shared__ bf16_t an[16][C_ + 8];   // row stride 264*2=528B (16B aligned)

  {
    int r = tid >> 3, seg = (tid & 7) * 32;
    const float* src = X + (size_t)(row0 + r) * C_ + seg;
    float s = 0.f, s2 = 0.f;
#pragma unroll
    for (int c = 0; c < 32; ++c) {
      float v = src[c];
      xr[r][seg + c] = v;
      s += v; s2 += v * v;
    }
    red[r][tid & 7][0] = s;
    red[r][tid & 7][1] = s2;
  }
  __syncthreads();
  if (tid < 16) {
    float s = 0.f, s2 = 0.f;
#pragma unroll
    for (int p = 0; p < 8; ++p) { s += red[tid][p][0]; s2 += red[tid][p][1]; }
    float mean = s * (1.0f / C_);
    float var  = s2 * (1.0f / C_) - mean * mean;
    mu[tid] = mean;
    rs[tid] = rsqrtf(var + 1e-5f);
  }
  __syncthreads();
  {
    int r = tid >> 3, seg = (tid & 7) * 32;
    float mean = mu[r], rstd = rs[r];
#pragma unroll
    for (int c = 0; c < 32; ++c) {
      int cc = seg + c;
      an[r][cc] = (bf16_t)((xr[r][cc] - mean) * rstd * gamma[cc] + beta[cc]);
    }
  }
  __syncthreads();

  const bf16_t* arow = an[lane & 15];
  const bf16_t* wrow[4];
  int nn[4];
#pragma unroll
  for (int t = 0; t < 4; ++t) {
    nn[t]   = col0 + t * 16 + (lane & 15);
    wrow[t] = Wt + (size_t)nn[t] * C_;
  }
  v8f acc[4];
#pragma unroll
  for (int t = 0; t < 4; ++t) acc[t] = (v8f){0.f,0.f,0.f,0.f,0.f,0.f,0.f,0.f};

#pragma unroll
  for (int kk = 0; kk < C_; kk += 32) {
    v16bf a = frag_ld(arow, kk, half);
#pragma unroll
    for (int t = 0; t < 4; ++t)
      acc[t] = wmma_bf16(a, frag_ld(wrow[t], kk, half), acc[t]);
  }

#pragma unroll
  for (int t = 0; t < 4; ++t) {
    float bb = bias[nn[t]];
#pragma unroll
    for (int i = 0; i < 8; ++i) {
      int r = i + half * 8;
      float v = (acc[t][i] + bb) * scale;
      if (gelu) v = 0.5f * v * (1.0f + erff(v * 0.70710678118654752f));
      out[(size_t)(row0 + r) * N + nn[t]] = (bf16_t)v;
    }
  }
}

// =====================================================================
// Kernel 2: V = sum_n LN(y[n]) @ Wv[n] + sum_n bv[n]
// Output HEAD-TRANSPOSED:  vt[((b*H + h)*D + d)*T2 + t]   (bf16)
// block=128 (4 waves x 16x64 strips = all 256 cols), grid = (R/16)
// =====================================================================
__global__ void cab_lnv_gemm(const float* __restrict__ Y,       // [2,R,256]
                             const float* __restrict__ gamma,   // [2,256]
                             const float* __restrict__ beta,    // [2,256]
                             const bf16_t* __restrict__ Wvt,    // [2][256,256] transposed
                             const float* __restrict__ bv,      // [2,256]
                             bf16_t* __restrict__ vt) {
  const int tid  = threadIdx.x;
  const int lane = tid & 31;
  const int wave = tid >> 5;
  const int half = lane >> 4;
  const int row0 = blockIdx.x * 16;
  const int col0 = wave * 64;

  __shared__ float  xr[16][C_];
  __shared__ float  red[16][8][2];
  __shared__ float  mu[16], rs[16];
  __shared__ bf16_t an[2][16][C_ + 8];

  for (int nb = 0; nb < 2; ++nb) {
    {
      int r = tid >> 3, seg = (tid & 7) * 32;
      const float* src = Y + (size_t)nb * R_ * C_ + (size_t)(row0 + r) * C_ + seg;
      float s = 0.f, s2 = 0.f;
#pragma unroll
      for (int c = 0; c < 32; ++c) {
        float v = src[c];
        xr[r][seg + c] = v;
        s += v; s2 += v * v;
      }
      red[r][tid & 7][0] = s;
      red[r][tid & 7][1] = s2;
    }
    __syncthreads();
    if (tid < 16) {
      float s = 0.f, s2 = 0.f;
#pragma unroll
      for (int p = 0; p < 8; ++p) { s += red[tid][p][0]; s2 += red[tid][p][1]; }
      float mean = s * (1.0f / C_);
      float var  = s2 * (1.0f / C_) - mean * mean;
      mu[tid] = mean;
      rs[tid] = rsqrtf(var + 1e-5f);
    }
    __syncthreads();
    {
      int r = tid >> 3, seg = (tid & 7) * 32;
      float mean = mu[r], rstd = rs[r];
#pragma unroll
      for (int c = 0; c < 32; ++c) {
        int cc = seg + c;
        an[nb][r][cc] =
            (bf16_t)((xr[r][cc] - mean) * rstd * gamma[nb * C_ + cc] + beta[nb * C_ + cc]);
      }
    }
    __syncthreads();
  }

  int nn[4];
#pragma unroll
  for (int t = 0; t < 4; ++t) nn[t] = col0 + t * 16 + (lane & 15);

  v8f acc[4];
#pragma unroll
  for (int t = 0; t < 4; ++t) acc[t] = (v8f){0.f,0.f,0.f,0.f,0.f,0.f,0.f,0.f};

#pragma unroll
  for (int nb = 0; nb < 2; ++nb) {
    const bf16_t* arow = an[nb][lane & 15];
    const bf16_t* wbase = Wvt + (size_t)nb * C_ * C_;
#pragma unroll
    for (int kk = 0; kk < C_; kk += 32) {
      v16bf a = frag_ld(arow, kk, half);
#pragma unroll
      for (int t = 0; t < 4; ++t)
        acc[t] = wmma_bf16(a, frag_ld(wbase + (size_t)nn[t] * C_, kk, half), acc[t]);
    }
  }

#pragma unroll
  for (int t = 0; t < 4; ++t) {
    float bb = bv[nn[t]] + bv[C_ + nn[t]];
    const int h = nn[t] >> 5, d = nn[t] & 31;
#pragma unroll
    for (int i = 0; i < 8; ++i) {
      int r = row0 + i + half * 8;
      int b = r >> 11, tt = r & (T2_ - 1);
      vt[((size_t)(b * H_ + h) * D_ + d) * T2_ + tt] = (bf16_t)(acc[t][i] + bb);
    }
  }
}

// =====================================================================
// Kernel 3: streaming (flash) cross-attention, one wave per 16 q-rows.
// Decay multiplies the NUMERATOR only (reference applies it post-softmax).
// grid = (T1/16, H, B), block = 32.
// =====================================================================
#define PROW 40   // pbuf row stride (80B, 16B aligned)
__global__ void cab_attn(const bf16_t* __restrict__ qbuf,  // [R,256]
                         const bf16_t* __restrict__ kbuf,  // [R,256]
                         const bf16_t* __restrict__ vt,    // [B,H,D,T2]
                         const int* __restrict__ mask,     // [B,1,T1,T2]
                         const float* __restrict__ dist,   // [B,1,T1,T2]
                         bf16_t* __restrict__ obuf) {      // [R,256]
  const int lane = threadIdx.x;
  const int q0 = blockIdx.x * 16;
  const int h  = blockIdx.y;
  const int b  = blockIdx.z;
  const int m    = lane & 15;
  const int half = lane >> 4;

  __shared__ bf16_t pbuf[16 * PROW];

  const v16bf a_q =
      frag_ld(qbuf + (size_t)(b * T1_ + q0 + m) * C_ + h * D_, 0, half);

  float rowm[8], rowl[8];
#pragma unroll
  for (int i = 0; i < 8; ++i) { rowm[i] = -1e30f; rowl[i] = 0.f; }
  v8f acc0 = {0.f,0.f,0.f,0.f,0.f,0.f,0.f,0.f};
  v8f acc1 = {0.f,0.f,0.f,0.f,0.f,0.f,0.f,0.f};

  const bf16_t* vrow0 = vt + ((size_t)(b * H_ + h) * D_ +  0 + m) * T2_;
  const bf16_t* vrow1 = vt + ((size_t)(b * H_ + h) * D_ + 16 + m) * T2_;

  for (int j0 = 0; j0 < T2_; j0 += 32) {
    // ---- scores: two 16x16 WMMA tiles over 32 keys ----
    v8f s[2];
#pragma unroll
    for (int sub = 0; sub < 2; ++sub) {
      v16bf bk = frag_ld(
          kbuf + (size_t)(b * T2_ + j0 + sub * 16 + m) * C_ + h * D_, 0, half);
      v8f z = {0.f,0.f,0.f,0.f,0.f,0.f,0.f,0.f};
      s[sub] = wmma_bf16(a_q, bk, z);
    }
    // ---- mask + decay ----
    float dec[2][8];
#pragma unroll
    for (int sub = 0; sub < 2; ++sub) {
#pragma unroll
      for (int i = 0; i < 8; ++i) {
        int r = i + half * 8;
        size_t idx = (size_t)(b * T1_ + q0 + r) * T2_ + (j0 + sub * 16 + m);
        float sv = s[sub][i];
        if (mask[idx] == 0) sv = -1e9f;
        s[sub][i] = sv;
        float dd = dist[idx] * 2.0f;              // dist/GAMMA, GAMMA=0.5
        dec[sub][i] = __expf(-dd * dd);
      }
    }
    // ---- online softmax (row lives in 16 lanes of one half) ----
#pragma unroll
    for (int i = 0; i < 8; ++i) {
      float mx = fmaxf(s[0][i], s[1][i]);
#pragma unroll
      for (int off = 8; off > 0; off >>= 1) mx = fmaxf(mx, __shfl_xor(mx, off, 32));
      float mnew  = fmaxf(rowm[i], mx);
      float alpha = __expf(rowm[i] - mnew);
      float p0 = __expf(s[0][i] - mnew);
      float p1 = __expf(s[1][i] - mnew);
      float psum = p0 + p1;                       // denominator: NO decay
#pragma unroll
      for (int off = 8; off > 0; off >>= 1) psum += __shfl_xor(psum, off, 32);
      rowl[i] = rowl[i] * alpha + psum;
      rowm[i] = mnew;
      acc0[i] *= alpha;
      acc1[i] *= alpha;
      int r = i + half * 8;
      pbuf[r * PROW +  0 + m] = (bf16_t)(p0 * dec[0][i]);
      pbuf[r * PROW + 16 + m] = (bf16_t)(p1 * dec[1][i]);
    }
    // D-layout -> A-layout via LDS; wave-local so a dscnt wait suffices
    asm volatile("s_wait_dscnt 0" ::: "memory");
    v16bf a_p = frag_ld(pbuf + m * PROW, 0, half);
    // ---- accumulate P @ V (contiguous-key b128 fragments per d-tile) ----
    acc0 = wmma_bf16(a_p, frag_ld(vrow0, j0, half), acc0);
    acc1 = wmma_bf16(a_p, frag_ld(vrow1, j0, half), acc1);
    asm volatile("s_wait_dscnt 0" ::: "memory");  // reads done before next store
  }

#pragma unroll
  for (int i = 0; i < 8; ++i) {
    int r = i + half * 8;
    float rl = 1.0f / rowl[i];
    size_t o = (size_t)(b * T1_ + q0 + r) * C_ + h * D_;
    obuf[o +  0 + m] = (bf16_t)(acc0[i] * rl);
    obuf[o + 16 + m] = (bf16_t)(acc1[i] * rl);
  }
}

// =====================================================================
// Kernel 4: out[r,n] = Res[r,n] + (A@W)[r,n] + bias[n]
// A bf16 [R,K], Wt bf16 [256,K] (transposed), out f32 [R,256].
// block=128 (4 waves x 16x64 strips = 256 cols), grid = (R/16)
// =====================================================================
__global__ void cab_gemm_res(const bf16_t* __restrict__ A,
                             const bf16_t* __restrict__ Wt,
                             const float* __restrict__ bias,
                             const float* __restrict__ Res,
                             float* __restrict__ out, int K) {
  const int tid  = threadIdx.x;
  const int lane = tid & 31;
  const int wave = tid >> 5;
  const int half = lane >> 4;
  const int row0 = blockIdx.x * 16;
  const int col0 = wave * 64;

  const bf16_t* arow = A + (size_t)(row0 + (lane & 15)) * K;
  const bf16_t* wrow[4];
  int nn[4];
#pragma unroll
  for (int t = 0; t < 4; ++t) {
    nn[t]   = col0 + t * 16 + (lane & 15);
    wrow[t] = Wt + (size_t)nn[t] * K;
  }
  v8f acc[4];
#pragma unroll
  for (int t = 0; t < 4; ++t) acc[t] = (v8f){0.f,0.f,0.f,0.f,0.f,0.f,0.f,0.f};

#pragma unroll 2
  for (int kk = 0; kk < K; kk += 32) {
    v16bf a = frag_ld(arow, kk, half);
#pragma unroll
    for (int t = 0; t < 4; ++t)
      acc[t] = wmma_bf16(a, frag_ld(wrow[t], kk, half), acc[t]);
  }

#pragma unroll
  for (int t = 0; t < 4; ++t) {
    float bb = bias[nn[t]];
#pragma unroll
    for (int i = 0; i < 8; ++i) {
      int r = i + half * 8;
      size_t o = (size_t)(row0 + r) * C_ + nn[t];
      out[o] = Res[o] + acc[t][i] + bb;
    }
  }
}

// =====================================================================
extern "C" void kernel_launch(void* const* d_in, const int* in_sizes, int n_in,
                              void* d_out, int out_size, void* d_ws, size_t ws_size,
                              hipStream_t stream) {
  const float* x_q   = (const float*)d_in[0];
  const float* x_r   = (const float*)d_in[1];
  const float* y     = (const float*)d_in[2];
  const int*   mask  = (const int*)d_in[3];
  const float* dist  = (const float*)d_in[4];
  const float* Wq    = (const float*)d_in[5];
  const float* bq    = (const float*)d_in[6];
  const float* Wk    = (const float*)d_in[7];
  const float* bk    = (const float*)d_in[8];
  const float* Wv    = (const float*)d_in[9];
  const float* bv    = (const float*)d_in[10];
  const float* Wp    = (const float*)d_in[11];
  const float* bp    = (const float*)d_in[12];
  const float* ln1_g = (const float*)d_in[13];
  const float* ln1_b = (const float*)d_in[14];
  const float* ln2_g = (const float*)d_in[15];
  const float* ln2_b = (const float*)d_in[16];
  const float* lnb_g = (const float*)d_in[17];
  const float* lnb_b = (const float*)d_in[18];
  const float* ln3_g = (const float*)d_in[19];
  const float* ln3_b = (const float*)d_in[20];
  const float* Wm1   = (const float*)d_in[21];
  const float* bm1   = (const float*)d_in[22];
  const float* Wm2   = (const float*)d_in[23];
  const float* bm2   = (const float*)d_in[24];

  char* ws = (char*)d_ws;
  bf16_t* qb   = (bf16_t*)ws;  ws += (size_t)R_ * C_ * 2;
  bf16_t* kb   = (bf16_t*)ws;  ws += (size_t)R_ * C_ * 2;
  bf16_t* vtb  = (bf16_t*)ws;  ws += (size_t)B_ * H_ * D_ * T2_ * 2;
  bf16_t* ob   = (bf16_t*)ws;  ws += (size_t)R_ * C_ * 2;
  float*  xb   = (float*)ws;   ws += (size_t)R_ * C_ * 4;
  bf16_t* mb   = (bf16_t*)ws;  ws += (size_t)R_ * 4 * C_ * 2;
  bf16_t* wqt  = (bf16_t*)ws;  ws += (size_t)C_ * C_ * 2;
  bf16_t* wkt  = (bf16_t*)ws;  ws += (size_t)C_ * C_ * 2;
  bf16_t* wvt  = (bf16_t*)ws;  ws += (size_t)2 * C_ * C_ * 2;
  bf16_t* wpt  = (bf16_t*)ws;  ws += (size_t)C_ * C_ * 2;
  bf16_t* wm1t = (bf16_t*)ws;  ws += (size_t)C_ * 4 * C_ * 2;
  bf16_t* wm2t = (bf16_t*)ws;  ws += (size_t)4 * C_ * C_ * 2;

  // ---- weight transpose+convert passes ----
  const int sq = C_ * C_;
  cab_w2bf_t<<<dim3((sq + 255) / 256), 256, 0, stream>>>(Wq, wqt, C_, C_);
  cab_w2bf_t<<<dim3((sq + 255) / 256), 256, 0, stream>>>(Wk, wkt, C_, C_);
  cab_w2bf_t<<<dim3((sq + 255) / 256), 256, 0, stream>>>(Wv, wvt, C_, C_);
  cab_w2bf_t<<<dim3((sq + 255) / 256), 256, 0, stream>>>(Wv + sq, wvt + sq, C_, C_);
  cab_w2bf_t<<<dim3((sq + 255) / 256), 256, 0, stream>>>(Wp, wpt, C_, C_);
  cab_w2bf_t<<<dim3((4 * sq + 255) / 256), 256, 0, stream>>>(Wm1, wm1t, C_, 4 * C_);
  cab_w2bf_t<<<dim3((4 * sq + 255) / 256), 256, 0, stream>>>(Wm2, wm2t, 4 * C_, C_);

  const float qscale = 0.17677669529663688f;  // 1/sqrt(32)

  // Q = LN1(x_q)@Wq + bq (pre-scaled); K = LN2(x_r)@Wk + bk
  cab_ln_gemm<<<dim3(R_ / 16, 1), 128, 0, stream>>>(
      x_q, ln1_g, ln1_b, wqt, bq, qb, C_, qscale, 0);
  cab_ln_gemm<<<dim3(R_ / 16, 1), 128, 0, stream>>>(
      x_r, ln2_g, ln2_b, wkt, bk, kb, C_, 1.0f, 0);
  // V (head-transposed) = sum_n LN_n(y_n)@Wv_n + sum bv
  cab_lnv_gemm<<<dim3(R_ / 16), 128, 0, stream>>>(
      y, lnb_g, lnb_b, wvt, bv, vtb);
  // streaming attention
  cab_attn<<<dim3(T1_ / 16, H_, B_), 32, 0, stream>>>(qb, kb, vtb, mask, dist, ob);
  // x = x_q + attn_out @ Wp + bp
  cab_gemm_res<<<dim3(R_ / 16), 128, 0, stream>>>(ob, wpt, bp, x_q, xb, C_);
  // m = gelu(LN3(x)@Wm1 + bm1)
  cab_ln_gemm<<<dim3(R_ / 16, 4), 128, 0, stream>>>(
      xb, ln3_g, ln3_b, wm1t, bm1, mb, 4 * C_, 1.0f, 1);
  // out = x + m @ Wm2 + bm2
  cab_gemm_res<<<dim3(R_ / 16), 128, 0, stream>>>(
      mb, wm2t, bm2, xb, (float*)d_out, 4 * C_);
}
